// VectorQuantizer_33672543600894
// MI455X (gfx1250) — matrix operand order repaired and checked by reading, not compiled, not run on hardware
//
#include <hip/hip_runtime.h>

// ---------------------------------------------------------------------------
// VQ-VAE vector quantizer for MI455X (gfx1250, wave32, WMMA + async-LDS).
//   z  : [65536, 256] f32,  emb : [8192, 256] f32
//   out: [zq 65536*256 | idx 65536 (stored as float) | loss 1]  (f32)
//
// Compute-bound roofline: 2*N*K*D = 275 GFLOP; HBM floor (~6us) would need
// ~46 PFLOPS, so the GEMM precision path decides runtime. We use split-bf16
// (zhi*ehi + zhi*elo + zlo*ehi, f32 accumulate) on v_wmma_f32_16x16x32_bf16:
// ~16-bit effective mantissa (argmin-safe for this distribution), ~2.7x the
// native f32-WMMA (16x16x4) path.
//
// Fast path: emb pre-split once into ws as per-chunk LDS fragment images;
// main kernel double-buffers chunks into LDS with GLOBAL_LOAD_ASYNC_TO_LDS
// (ASYNCcnt), fully overlapping staging with the WMMA stream. Fallback path
// (ws too small): fused in-loop conversion.
// ---------------------------------------------------------------------------

typedef __attribute__((ext_vector_type(16))) __bf16        v16bf;
typedef __attribute__((ext_vector_type(8)))  float         v8f;
typedef __attribute__((ext_vector_type(8)))  unsigned int  v8u;

#define N_ROWS  65536
#define K_CODES 8192
#define DIM     256
#define M_TILE  128                    // rows per workgroup (8 waves x 16)
#define KC      32                     // codes per K-chunk
#define NCHUNK  (K_CODES / KC)         // 256
#define THREADS 256
#define NUM_WG  (N_ROWS / M_TILE)      // 512
#define CHUNK_DW 8192                  // dwords per chunk image (32 KB)
#define CHUNK_BYTES (CHUNK_DW * 4)

__device__ __forceinline__ unsigned int bf16_rne(float f) {
  unsigned int u = __builtin_bit_cast(unsigned int, f);
  unsigned int r = u + 0x7FFFu + ((u >> 16) & 1u);   // round-to-nearest-even
  return r >> 16;
}
__device__ __forceinline__ float bf16_to_f(unsigned int h) {
  return __builtin_bit_cast(float, h << 16);
}
__device__ __forceinline__ v16bf lds_frag(const unsigned int* base) {
  const uint4* p = (const uint4*)base;               // 2x ds_load_b128
  uint4 q0 = p[0], q1 = p[1];
  v8u u; u[0]=q0.x; u[1]=q0.y; u[2]=q0.z; u[3]=q0.w;
         u[4]=q1.x; u[5]=q1.y; u[6]=q1.z; u[7]=q1.w;
  return __builtin_bit_cast(v16bf, u);
}

// Congruent 32KB copy: 256 threads x 8 x b128, global(frag image) -> LDS.
// INST_OFFSET is added to both the global and LDS byte addresses (ISA 10.7).
__device__ __forceinline__ void issue_async_chunk(const unsigned int* frag,
                                                  int c, unsigned lds_base,
                                                  int t) {
  unsigned voff = (unsigned)(c * CHUNK_BYTES + t * 16);
  unsigned dst  = lds_base + (unsigned)(t * 16);
  const void* sb = (const void*)frag;
  asm volatile("global_load_async_to_lds_b128 %0, %1, %2 offset:0"
               :: "v"(dst), "v"(voff), "s"(sb) : "memory");
  asm volatile("global_load_async_to_lds_b128 %0, %1, %2 offset:4096"
               :: "v"(dst), "v"(voff), "s"(sb) : "memory");
  asm volatile("global_load_async_to_lds_b128 %0, %1, %2 offset:8192"
               :: "v"(dst), "v"(voff), "s"(sb) : "memory");
  asm volatile("global_load_async_to_lds_b128 %0, %1, %2 offset:12288"
               :: "v"(dst), "v"(voff), "s"(sb) : "memory");
  asm volatile("global_load_async_to_lds_b128 %0, %1, %2 offset:16384"
               :: "v"(dst), "v"(voff), "s"(sb) : "memory");
  asm volatile("global_load_async_to_lds_b128 %0, %1, %2 offset:20480"
               :: "v"(dst), "v"(voff), "s"(sb) : "memory");
  asm volatile("global_load_async_to_lds_b128 %0, %1, %2 offset:24576"
               :: "v"(dst), "v"(voff), "s"(sb) : "memory");
  asm volatile("global_load_async_to_lds_b128 %0, %1, %2 offset:28672"
               :: "v"(dst), "v"(voff), "s"(sb) : "memory");
}
__device__ __forceinline__ void wait_async0() {
  asm volatile("s_wait_asynccnt 0x0" ::: "memory");
}

// ---------------------------------------------------------------------------
// Pre-split kernel: emb -> per-chunk fragment images (split bf16) + ||e||^2.
// Image layout per chunk (dwords): [0..4095] hi frags, [4096..8191] lo frags,
// frag dword = ((s*8 + db)*32 + c16 + 16*((dd>>3)&1))*8 + (((dd&7) + 8*(dd>>4))>>1), half = dd&1
//   (s = code/16, c16 = code%16, db = d/32, dd = d%32; B: n=c16, k=dd in AMD table order)
// ---------------------------------------------------------------------------
__global__ __launch_bounds__(THREADS)
void vq_presplit(const float* __restrict__ emb,
                 unsigned int* __restrict__ frag,
                 float* __restrict__ enorm)
{
  const int t = threadIdx.x, lane = t & 31, w = t >> 5;
  const int c = blockIdx.x;                   // chunk id
  const int k0 = c * KC;
  unsigned int* base = frag + (size_t)c * CHUNK_DW;
  for (int i = w; i < KC; i += 8) {           // wave-owned codes: determinism
    const int s = i >> 4, c16 = i & 15;
    const float* erow = emb + (size_t)(k0 + i) * DIM;
    float sq = 0.f;
#pragma unroll
    for (int seg = 0; seg < 8; ++seg) {       // db = seg, dd = lane
      float e = erow[seg * 32 + lane];
      unsigned int hh = bf16_rne(e);
      unsigned int ll = bf16_rne(e - bf16_to_f(hh));
      int dw = ((s * 8 + seg) * 32 + (c16 + 16 * ((lane >> 3) & 1))) * 8 + (((lane & 7) + 8 * (lane >> 4)) >> 1);
      ((unsigned short*)&base[dw])[lane & 1]        = (unsigned short)hh;
      ((unsigned short*)&base[4096 + dw])[lane & 1] = (unsigned short)ll;
      sq += e * e;
    }
#pragma unroll
    for (int m = 16; m >= 1; m >>= 1) sq += __shfl_xor(sq, m, 32);
    if (lane == 0) enorm[k0 + i] = sq;
  }
}

// ---------------------------------------------------------------------------
// Main kernel, fast path: async double-buffered staging + WMMA K-loop.
// ---------------------------------------------------------------------------
__global__ __launch_bounds__(THREADS)
void vq_main_async(const float* __restrict__ z, const float* __restrict__ emb,
                   const unsigned int* __restrict__ frag,
                   const float* __restrict__ enorm,
                   float* __restrict__ out_zq, float* __restrict__ out_idx,
                   float* __restrict__ ws_partial)
{
  __shared__ __align__(16) unsigned int s_buf[2][CHUNK_DW];   // 64 KB
  __shared__ int   s_idx[M_TILE];
  __shared__ float s_red[8];

  const int t    = threadIdx.x;
  const int lane = t & 31;
  const int w    = t >> 5;
  const int lrow = lane & 15;
  const int lhi  = lane >> 4;
  const int row0 = blockIdx.x * M_TILE;

  // ---- A fragments (wave's 16 z-rows, split bf16 hi/lo) resident in VGPRs.
  // 16-bit A 16x32 layout: lane holds M=lane%16; VGPR v holds K pair at
  // d = 32*db + 2*(v&3) + 16*(v>>2) + 8*(lane>>4).
  v16bf a_hi[8], a_lo[8];
  {
    const float* zrow = z + (size_t)(row0 + w * 16 + lrow) * DIM;
#pragma unroll
    for (int db = 0; db < 8; ++db) {
      v8u uh, ul;
#pragma unroll
      for (int v = 0; v < 8; ++v) {
        int d = 32 * db + ((v & 3) * 2) + 16 * (v >> 2) + 8 * lhi;
        float2 x = *(const float2*)(zrow + d);
        unsigned int h0 = bf16_rne(x.x);
        unsigned int h1 = bf16_rne(x.y);
        unsigned int l0 = bf16_rne(x.x - bf16_to_f(h0));
        unsigned int l1 = bf16_rne(x.y - bf16_to_f(h1));
        uh[v] = h0 | (h1 << 16);
        ul[v] = l0 | (l1 << 16);
      }
      a_hi[db] = __builtin_bit_cast(v16bf, uh);
      a_lo[db] = __builtin_bit_cast(v16bf, ul);
    }
  }

  float best[8];
  int   bidx[8];
#pragma unroll
  for (int j = 0; j < 8; ++j) { best[j] = 3.4e38f; bidx[j] = 0; }

  // Generic shared pointer: addr[31:0] is the wave-relative LDS byte offset.
  const unsigned lds0 = (unsigned)(unsigned long long)(const void*)&s_buf[0][0];

  issue_async_chunk(frag, 0, lds0, t);                 // prologue

  for (int c = 0; c < NCHUNK; ++c) {
    const int cur = c & 1;
    wait_async0();            // my buf[cur] writes landed
    __syncthreads();          // everyone's landed; buf[cur^1] reads retired
    if (c + 1 < NCHUNK)
      issue_async_chunk(frag, c + 1, lds0 + (unsigned)((cur ^ 1) * CHUNK_BYTES), t);

    const unsigned int* bh = &s_buf[cur][0];
    const unsigned int* bl = &s_buf[cur][4096];

    v8f acc0 = {}; v8f acc1 = {};
#pragma unroll
    for (int db = 0; db < 8; ++db) {
      v16bf b0h = lds_frag(&bh[((0 * 8 + db) * 32 + lane) * 8]);
      v16bf b1h = lds_frag(&bh[((1 * 8 + db) * 32 + lane) * 8]);
      v16bf b0l = lds_frag(&bl[((0 * 8 + db) * 32 + lane) * 8]);
      v16bf b1l = lds_frag(&bl[((1 * 8 + db) * 32 + lane) * 8]);
      acc0 = __builtin_amdgcn_wmma_f32_16x16x32_bf16(false, a_hi[db], false, b0h, (short)0, acc0, false, false);
      acc1 = __builtin_amdgcn_wmma_f32_16x16x32_bf16(false, a_hi[db], false, b1h, (short)0, acc1, false, false);
      acc0 = __builtin_amdgcn_wmma_f32_16x16x32_bf16(false, a_hi[db], false, b0l, (short)0, acc0, false, false);
      acc1 = __builtin_amdgcn_wmma_f32_16x16x32_bf16(false, a_hi[db], false, b1l, (short)0, acc1, false, false);
      acc0 = __builtin_amdgcn_wmma_f32_16x16x32_bf16(false, a_lo[db], false, b0h, (short)0, acc0, false, false);
      acc1 = __builtin_amdgcn_wmma_f32_16x16x32_bf16(false, a_lo[db], false, b1h, (short)0, acc1, false, false);
    }

    // d = ||e||^2 - 2 z.e  (||z||^2 constant per row: argmin-invariant)
    const int k0 = c * KC;
    float en0 = enorm[k0 + lrow];
    float en1 = enorm[k0 + 16 + lrow];
    int kk0 = k0 + lrow, kk1 = k0 + 16 + lrow;
#pragma unroll
    for (int j = 0; j < 8; ++j) {
      float d0 = en0 - 2.0f * acc0[j];
      float d1 = en1 - 2.0f * acc1[j];
      if (d0 < best[j]) { best[j] = d0; bidx[j] = kk0; }   // strict < keeps
      if (d1 < best[j]) { best[j] = d1; bidx[j] = kk1; }   // first occurrence
    }
  }

  // Cross-lane argmin over the 16 lanes of a row (first-index tie-break).
#pragma unroll
  for (int j = 0; j < 8; ++j) {
    float v = best[j]; int bi = bidx[j];
#pragma unroll
    for (int m = 1; m <= 8; m <<= 1) {
      float v2 = __shfl_xor(v, m, 32);
      int   b2 = __shfl_xor(bi, m, 32);
      if (v2 < v || (v2 == v && b2 < bi)) { v = v2; bi = b2; }
    }
    int mlocal = w * 16 + j + 8 * lhi;
    if (lrow == 0) {
      s_idx[mlocal] = bi;
      out_idx[row0 + mlocal] = (float)bi;
    }
  }
  __syncthreads();

  // ---- zq gather + loss partial (fused) ----
  float lsum = 0.f;
  for (int i = 0; i < M_TILE; ++i) {
    int kbest = s_idx[i];
    float e  = emb[(size_t)kbest * DIM + t];
    float zv = z[(size_t)(row0 + i) * DIM + t];
    out_zq[(size_t)(row0 + i) * DIM + t] = e;
    float df = e - zv;
    lsum += df * df;
  }
#pragma unroll
  for (int m = 16; m >= 1; m >>= 1) lsum += __shfl_xor(lsum, m, 32);
  if (lane == 0) s_red[w] = lsum;
  __syncthreads();
  if (t == 0) {
    float tot = 0.f;
#pragma unroll
    for (int i2 = 0; i2 < 8; ++i2) tot += s_red[i2];
    ws_partial[blockIdx.x] = tot;
  }
}

// ---------------------------------------------------------------------------
// Fallback path (ws too small for split table): fused in-loop staging.
// ---------------------------------------------------------------------------
__global__ __launch_bounds__(THREADS)
void vq_main_fused(const float* __restrict__ z, const float* __restrict__ emb,
                   float* __restrict__ out_zq, float* __restrict__ out_idx,
                   float* __restrict__ ws_partial)
{
  __shared__ __align__(16) unsigned int s_bhi[4096];
  __shared__ __align__(16) unsigned int s_blo[4096];
  __shared__ float s_enorm[KC];
  __shared__ int   s_idx[M_TILE];
  __shared__ float s_red[8];

  const int t    = threadIdx.x;
  const int lane = t & 31;
  const int w    = t >> 5;
  const int lrow = lane & 15;
  const int lhi  = lane >> 4;
  const int row0 = blockIdx.x * M_TILE;

  v16bf a_hi[8], a_lo[8];
  {
    const float* zrow = z + (size_t)(row0 + w * 16 + lrow) * DIM;
#pragma unroll
    for (int db = 0; db < 8; ++db) {
      v8u uh, ul;
#pragma unroll
      for (int v = 0; v < 8; ++v) {
        int d = 32 * db + ((v & 3) * 2) + 16 * (v >> 2) + 8 * lhi;
        float2 x = *(const float2*)(zrow + d);
        unsigned int h0 = bf16_rne(x.x);
        unsigned int h1 = bf16_rne(x.y);
        unsigned int l0 = bf16_rne(x.x - bf16_to_f(h0));
        unsigned int l1 = bf16_rne(x.y - bf16_to_f(h1));
        uh[v] = h0 | (h1 << 16);
        ul[v] = l0 | (l1 << 16);
      }
      a_hi[db] = __builtin_bit_cast(v16bf, uh);
      a_lo[db] = __builtin_bit_cast(v16bf, ul);
    }
  }

  float best[8];
  int   bidx[8];
#pragma unroll
  for (int j = 0; j < 8; ++j) { best[j] = 3.4e38f; bidx[j] = 0; }

  for (int k0 = 0; k0 < K_CODES; k0 += KC) {
    __syncthreads();
    for (int i = w; i < KC; i += 8) {
      const int s = i >> 4, c16 = i & 15;
      const float* erow = emb + (size_t)(k0 + i) * DIM;
      float sq = 0.f;
#pragma unroll
      for (int seg = 0; seg < 8; ++seg) {
        float e = erow[seg * 32 + lane];
        unsigned int hh = bf16_rne(e);
        unsigned int ll = bf16_rne(e - bf16_to_f(hh));
        int dw = ((s * 8 + seg) * 32 + (c16 + 16 * ((lane >> 3) & 1))) * 8 + (((lane & 7) + 8 * (lane >> 4)) >> 1);
        ((unsigned short*)&s_bhi[dw])[lane & 1] = (unsigned short)hh;
        ((unsigned short*)&s_blo[dw])[lane & 1] = (unsigned short)ll;
        sq += e * e;
      }
#pragma unroll
      for (int m = 16; m >= 1; m >>= 1) sq += __shfl_xor(sq, m, 32);
      if (lane == 0) s_enorm[i] = sq;
    }
    __syncthreads();

    v8f acc0 = {}; v8f acc1 = {};
#pragma unroll
    for (int db = 0; db < 8; ++db) {
      v16bf b0h = lds_frag(&s_bhi[((0 * 8 + db) * 32 + lane) * 8]);
      v16bf b1h = lds_frag(&s_bhi[((1 * 8 + db) * 32 + lane) * 8]);
      v16bf b0l = lds_frag(&s_blo[((0 * 8 + db) * 32 + lane) * 8]);
      v16bf b1l = lds_frag(&s_blo[((1 * 8 + db) * 32 + lane) * 8]);
      acc0 = __builtin_amdgcn_wmma_f32_16x16x32_bf16(false, a_hi[db], false, b0h, (short)0, acc0, false, false);
      acc1 = __builtin_amdgcn_wmma_f32_16x16x32_bf16(false, a_hi[db], false, b1h, (short)0, acc1, false, false);
      acc0 = __builtin_amdgcn_wmma_f32_16x16x32_bf16(false, a_hi[db], false, b0l, (short)0, acc0, false, false);
      acc1 = __builtin_amdgcn_wmma_f32_16x16x32_bf16(false, a_hi[db], false, b1l, (short)0, acc1, false, false);
      acc0 = __builtin_amdgcn_wmma_f32_16x16x32_bf16(false, a_lo[db], false, b0h, (short)0, acc0, false, false);
      acc1 = __builtin_amdgcn_wmma_f32_16x16x32_bf16(false, a_lo[db], false, b1h, (short)0, acc1, false, false);
    }

    float en0 = s_enorm[lrow];
    float en1 = s_enorm[16 + lrow];
    int kk0 = k0 + lrow, kk1 = k0 + 16 + lrow;
#pragma unroll
    for (int j = 0; j < 8; ++j) {
      float d0 = en0 - 2.0f * acc0[j];
      float d1 = en1 - 2.0f * acc1[j];
      if (d0 < best[j]) { best[j] = d0; bidx[j] = kk0; }
      if (d1 < best[j]) { best[j] = d1; bidx[j] = kk1; }
    }
  }

#pragma unroll
  for (int j = 0; j < 8; ++j) {
    float v = best[j]; int bi = bidx[j];
#pragma unroll
    for (int m = 1; m <= 8; m <<= 1) {
      float v2 = __shfl_xor(v, m, 32);
      int   b2 = __shfl_xor(bi, m, 32);
      if (v2 < v || (v2 == v && b2 < bi)) { v = v2; bi = b2; }
    }
    int mlocal = w * 16 + j + 8 * lhi;
    if (lrow == 0) {
      s_idx[mlocal] = bi;
      out_idx[row0 + mlocal] = (float)bi;
    }
  }
  __syncthreads();

  float lsum = 0.f;
  for (int i = 0; i < M_TILE; ++i) {
    int kbest = s_idx[i];
    float e  = emb[(size_t)kbest * DIM + t];
    float zv = z[(size_t)(row0 + i) * DIM + t];
    out_zq[(size_t)(row0 + i) * DIM + t] = e;
    float df = e - zv;
    lsum += df * df;
  }
#pragma unroll
  for (int m = 16; m >= 1; m >>= 1) lsum += __shfl_xor(lsum, m, 32);
  if (lane == 0) s_red[w] = lsum;
  __syncthreads();
  if (t == 0) {
    float tot = 0.f;
#pragma unroll
    for (int i2 = 0; i2 < 8; ++i2) tot += s_red[i2];
    ws_partial[blockIdx.x] = tot;
  }
}

__global__ void vq_finalize(const float* __restrict__ ws_partial,
                            float* __restrict__ out_loss)
{
  if (threadIdx.x == 0) {
    float s = 0.f;
    for (int i = 0; i < NUM_WG; ++i) s += ws_partial[i];
    out_loss[0] = 1.25f * s / (float)(N_ROWS * DIM);  // (1 + 0.25) * MSE
  }
}

extern "C" void kernel_launch(void* const* d_in, const int* in_sizes, int n_in,
                              void* d_out, int out_size, void* d_ws, size_t ws_size,
                              hipStream_t stream)
{
  const float* z   = (const float*)d_in[0];
  const float* emb = (const float*)d_in[1];
  float* out      = (float*)d_out;
  float* out_zq   = out;
  float* out_idx  = out + (size_t)N_ROWS * DIM;
  float* out_loss = out + (size_t)N_ROWS * DIM + N_ROWS;

  const size_t frag_bytes  = (size_t)NCHUNK * CHUNK_BYTES;       // 8 MB
  const size_t enorm_bytes = (size_t)K_CODES * sizeof(float);    // 32 KB
  const size_t need = frag_bytes + enorm_bytes + NUM_WG * sizeof(float);

  if (ws_size >= need) {
    unsigned int* frag  = (unsigned int*)d_ws;
    float* enorm        = (float*)((char*)d_ws + frag_bytes);
    float* partial      = (float*)((char*)d_ws + frag_bytes + enorm_bytes);
    vq_presplit<<<NCHUNK, THREADS, 0, stream>>>(emb, frag, enorm);
    vq_main_async<<<NUM_WG, THREADS, 0, stream>>>(z, emb, frag, enorm,
                                                  out_zq, out_idx, partial);
    vq_finalize<<<1, 1, 0, stream>>>(partial, out_loss);
  } else {
    float* partial = (float*)d_ws;
    vq_main_fused<<<NUM_WG, THREADS, 0, stream>>>(z, emb, out_zq, out_idx, partial);
    vq_finalize<<<1, 1, 0, stream>>>(partial, out_loss);
  }
}
